// GRUDModel_90941637525970
// MI455X (gfx1250) — compile-verified, matrix operand order checked by reading, not run on hardware
//
#include <hip/hip_runtime.h>
#include <hip/hip_bf16.h>
#include <math.h>

// GRU-D for MI455X (gfx1250, wave32, WMMA 16x16x32 bf16 + async-to-LDS)
// B=512, T=256, D=64, H=256

typedef __attribute__((ext_vector_type(16))) __bf16 v16bf;
typedef __attribute__((ext_vector_type(8)))  float  v8f;

#define B_   512
#define T_   256
#define D_   64
#define H_   256
#define KIN  192   // packed GEMM-2 K: [x_hat(64) | m(64) | delta(64)]
#define NOUT 1024  // packed GEMM-2 N: [z(256) r(256) ht(256) | gamma_h(256)]

__device__ __forceinline__ float sigmoidf_(float x) { return 1.0f / (1.0f + __expf(-x)); }

// 16-bit WMMA A/B fragment K mapping: lanes 0-15 hold K 0..7,16..23;
// lanes 16-31 hold K 8..15,24..31 (element j: j<8 -> +j, j>=8 -> +8+j).
__device__ __forceinline__ int kfrag_off(int hl, int j) {
    return hl * 8 + (j & 7) + ((j >> 3) << 4);
}
// Inverse: K-within-tile (0..31) -> packed fragment slot offset.
__device__ __forceinline__ int afrag_slot(int ktile, int m, int kk32) {
    int hl2 = (kk32 >> 3) & 1;
    int j   = (kk32 & 7) | (((kk32 >> 4) & 1) << 3);
    return (ktile * 32 + (hl2 << 4) + m) * 16 + j;
}

// ---------------------------------------------------------------------------
// Kernel P: pack all GEMM B-operands straight into per-lane fragment order.
// ---------------------------------------------------------------------------
__global__ void k_pack(const float* __restrict__ Wx, const float* __restrict__ Wm,
                       const float* __restrict__ Wgh, const float* __restrict__ Wh,
                       const float* __restrict__ Wgx,
                       __bf16* __restrict__ WcatF, __bf16* __restrict__ WhF,
                       __bf16* __restrict__ WgxF) {
    int i = blockIdx.x * blockDim.x + threadIdx.x;
    const int NC = 6 * 64 * 512;      // 196608
    const int NH = 3 * 8 * 16 * 512;  // 196608
    if (i < NC) {
        int s = i & 511, ft = i >> 9;
        int ktile = ft >> 6, ntile = ft & 63;
        int lane = s >> 4, j = s & 15;
        int kk = ktile * 32 + kfrag_off(lane >> 4, j);
        int n  = ntile * 16 + (lane & 15);
        float v = 0.0f;
        if (n < 768) {
            int g = n >> 8, h = n & 255;
            if (kk < 64)       v = Wx[(g * 64 + kk) * 256 + h];
            else if (kk < 128) v = Wm[(g * 64 + kk - 64) * 256 + h];
        } else if (kk >= 128) {
            v = Wgh[(kk - 128) * 256 + (n - 768)];
        }
        WcatF[i] = (__bf16)v;
    } else if (i < NC + NH) {
        int idx = i - NC;
        int s = idx & 511, ft = idx >> 9;
        int g = ft >> 7, rem = ft & 127;
        int ktile = rem >> 4, ntile = rem & 15;
        int lane = s >> 4, j = s & 15;
        int kk = ktile * 32 + kfrag_off(lane >> 4, j);
        int n  = ntile * 16 + (lane & 15);
        WhF[idx] = (__bf16)Wh[(g * 256 + kk) * 256 + n];
    } else if (i < NC + NH + 4096) {
        int idx = i - NC - NH;
        int s = idx & 511, ft = idx >> 9;
        int ktile = ft >> 2, ntile = ft & 3;
        int lane = s >> 4, j = s & 15;
        int kk = ktile * 32 + kfrag_off(lane >> 4, j);
        int n  = ntile * 16 + (lane & 15);
        WgxF[idx] = (__bf16)Wgx[kk * 64 + n];
    }
}

// ---------------------------------------------------------------------------
// Kernel A: x_prev carry scan (elementwise recurrence per (b,d) chain).
// ---------------------------------------------------------------------------
__global__ void k_scan(const float* __restrict__ X, const float* __restrict__ M,
                       float* __restrict__ xprev) {
    int tid = blockIdx.x * blockDim.x + threadIdx.x;   // 32768 = B*D
    int b = tid >> 6, d = tid & 63;
    float xp = 0.0f;
    size_t base = (size_t)b * T_ * D_ + d;
    for (int t = 0; t < T_; ++t) {
        size_t idx = base + (size_t)t * D_;
        xprev[idx] = xp;
        float m = M[idx];
        xp = m * X[idx] + (1.0f - m) * xp;
    }
}

// ---------------------------------------------------------------------------
// Kernel B: gamma_x GEMM [BT,64]x[64,64] + x_hat epilogue -> fragment-ordered
// XinF = [x_hat | m | delta] in bf16.
// ---------------------------------------------------------------------------
__global__ void k_gammax(const float* __restrict__ X, const float* __restrict__ M,
                         const float* __restrict__ Delta, const float* __restrict__ xmean,
                         const __bf16* __restrict__ WgxF, const float* __restrict__ bgx,
                         const float* __restrict__ xprev, __bf16* __restrict__ XinF) {
    int mt = (blockIdx.x * blockDim.x + threadIdx.x) >> 5;   // 8192 M-tiles
    int lane = threadIdx.x & 31;
    int hl = lane >> 4, ln = lane & 15;
    int row0 = mt * 16;
    int arow = row0 + ln;

    v8f acc[4] = {};
    #pragma unroll
    for (int k = 0; k < 2; ++k) {
        const float* ap = Delta + (size_t)arow * 64 + k * 32 + hl * 8;
        float4 f0 = *(const float4*)(ap);
        float4 f1 = *(const float4*)(ap + 4);
        float4 f2 = *(const float4*)(ap + 16);
        float4 f3 = *(const float4*)(ap + 20);
        v16bf a;
        a[0]=(__bf16)f0.x; a[1]=(__bf16)f0.y; a[2]=(__bf16)f0.z; a[3]=(__bf16)f0.w;
        a[4]=(__bf16)f1.x; a[5]=(__bf16)f1.y; a[6]=(__bf16)f1.z; a[7]=(__bf16)f1.w;
        a[8]=(__bf16)f2.x; a[9]=(__bf16)f2.y; a[10]=(__bf16)f2.z; a[11]=(__bf16)f2.w;
        a[12]=(__bf16)f3.x; a[13]=(__bf16)f3.y; a[14]=(__bf16)f3.z; a[15]=(__bf16)f3.w;
        #pragma unroll
        for (int nt = 0; nt < 4; ++nt) {
            v16bf bf = *(const v16bf*)(WgxF + (size_t)((k * 4 + nt) * 32 + lane) * 16);
            acc[nt] = __builtin_amdgcn_wmma_f32_16x16x32_bf16(
                false, a, false, bf, (short)0, acc[nt], false, false);
        }
    }
    #pragma unroll
    for (int nt = 0; nt < 4; ++nt) {
        int n = nt * 16 + ln;
        int kt0 = n >> 5, kk32 = n & 31;
        int hl2 = (kk32 >> 3) & 1;
        int j   = (kk32 & 7) | (((kk32 >> 4) & 1) << 3);
        #pragma unroll
        for (int i = 0; i < 8; ++i) {
            int rl = i + 8 * hl;              // row within M-tile
            int r  = row0 + rl;
            size_t idx = (size_t)r * 64 + n;
            float gx = __expf(-fmaxf(acc[nt][i] + bgx[n], 0.0f));
            float x = X[idx], mm = M[idx], xp = xprev[idx], dl = Delta[idx];
            float xh = mm * x + (1.0f - mm) * (gx * xp + (1.0f - gx) * xmean[n]);
            size_t base = ((size_t)mt * 6) * 512 + (size_t)((hl2 << 4) + rl) * 16 + j;
            XinF[base + (size_t)(kt0    ) * 512] = (__bf16)xh;   // col n
            XinF[base + (size_t)(kt0 + 2) * 512] = (__bf16)mm;   // col 64+n
            XinF[base + (size_t)(kt0 + 4) * 512] = (__bf16)dl;   // col 128+n
        }
    }
}

// ---------------------------------------------------------------------------
// Kernel C: fused GEMM [BT,192]x[192,1024] bf16 -> gate preacts + gamma_h.
// Time-major outputs: Aout[t][b][768], Gh[t][b][256]. Wave = 16x64 tile.
// ---------------------------------------------------------------------------
__global__ void k_biggemm(const __bf16* __restrict__ XinF, const __bf16* __restrict__ WcatF,
                          const float* __restrict__ bg, const float* __restrict__ bgh,
                          float* __restrict__ Aout, float* __restrict__ Gh) {
    int gw = (blockIdx.x * blockDim.x + threadIdx.x) >> 5;  // 131072 waves
    int lane = threadIdx.x & 31;
    int hl = lane >> 4, ln = lane & 15;
    int mt = gw >> 4;      // 8192 M-tiles
    int ng = gw & 15;      // 16 groups of 4 N-tiles
    int row0 = mt * 16;

    v8f acc[4] = {};
    #pragma unroll
    for (int k = 0; k < 6; ++k) {
        v16bf a = *(const v16bf*)(XinF + ((size_t)(mt * 6 + k) * 32 + lane) * 16);
        #pragma unroll
        for (int nt = 0; nt < 4; ++nt) {
            v16bf bf = *(const v16bf*)(WcatF + ((size_t)(k * 64 + ng * 4 + nt) * 32 + lane) * 16);
            acc[nt] = __builtin_amdgcn_wmma_f32_16x16x32_bf16(
                false, a, false, bf, (short)0, acc[nt], false, false);
        }
    }
    #pragma unroll
    for (int nt = 0; nt < 4; ++nt) {
        int n = (ng * 4 + nt) * 16 + ln;
        #pragma unroll
        for (int i = 0; i < 8; ++i) {
            int r = row0 + i + 8 * hl;   // r = b*T + t
            int b = r >> 8, t = r & 255; // T = 256
            if (n < 768) {
                Aout[((size_t)t * B_ + b) * 768 + n] = acc[nt][i] + bg[n];
            } else {
                Gh[((size_t)t * B_ + b) * 256 + (n - 768)] =
                    __expf(-fmaxf(acc[nt][i] + bgh[n - 768], 0.0f));
            }
        }
    }
}

// ---------------------------------------------------------------------------
// hd staged into LDS in A-fragment order (gamma_h read from LDS, async-filled).
// ---------------------------------------------------------------------------
__device__ __forceinline__ void hd_phase(const float* __restrict__ gh_s,
                                         const float* __restrict__ h_lds,
                                         __bf16* __restrict__ hdF, int tid) {
    #pragma unroll
    for (int j = 0; j < 4; ++j) {
        int f = tid * 4 + j;                 // flat (m,n) over 16x256
        int m = f >> 8, n = f & 255;
        float hd = gh_s[f] * h_lds[f];
        hdF[afrag_slot(n >> 5, m, n & 31)] = (__bf16)hd;
    }
}

// Async bulk copy: one B128 per lane, LDS dest address from the low 32 bits of
// the generic pointer (ISA 10.2: LDS aperture keeps the byte offset in [31:0]).
__device__ __forceinline__ void async_copy_b128(const void* gsrc, void* ldst) {
    unsigned dst = (unsigned)(size_t)ldst;
    asm volatile("global_load_async_to_lds_b128 %0, %1, off"
                 :: "v"(dst), "v"(gsrc) : "memory");
}
__device__ __forceinline__ void async_wait0() {
    asm volatile("s_wait_asynccnt 0x0" ::: "memory");
}

// ---------------------------------------------------------------------------
// Kernel R: serial recurrence. 32 WGs x 1024 threads (32 waves).
// Waves 0-15: z + h~ columns (wz+wt register-resident); waves 16-31: r columns.
// Per step: async gamma_h -> LDS overlapped with hoisted C-fragment loads,
// z-GEMM || r-GEMM, r-gating, h~-GEMM (r-waves prefetch t+1), h update.
// ---------------------------------------------------------------------------
__global__ __launch_bounds__(1024, 1)
void k_recur(const float* __restrict__ Aout, const float* __restrict__ Gh,
             const __bf16* __restrict__ WhF, const float* __restrict__ Wcls,
             const float* __restrict__ bcls, float* __restrict__ out) {
    __shared__ float  h_lds[16 * 256];       // fp32 hidden state
    __shared__ float  gh_s[16 * 256];        // gamma_h staging (async target)
    __shared__ __bf16 hdF[8 * 512];          // gamma_h*h, A-frag order
    __shared__ __bf16 rhdF[8 * 512];         // r*hd, A-frag order

    const int tid = threadIdx.x;
    const int wave = tid >> 5, lane = tid & 31;
    const int hl = lane >> 4, ln = lane & 15;
    const int b0 = blockIdx.x * 16;

    #pragma unroll
    for (int j = 0; j < 4; ++j) h_lds[tid * 4 + j] = 0.0f;

    if (wave < 16) {
        const int n0 = wave * 16;
        v16bf wz[8], wt[8];
        #pragma unroll
        for (int k = 0; k < 8; ++k) {
            wz[k] = *(const v16bf*)(WhF + ((size_t)((0 * 8 + k) * 16 + wave) * 32 + lane) * 16);
            wt[k] = *(const v16bf*)(WhF + ((size_t)((2 * 8 + k) * 16 + wave) * 32 + lane) * 16);
        }
        __syncthreads();
        for (int t = 0; t < T_; ++t) {
            // async gamma_h block -> LDS; C-fragments hoisted to overlap it
            async_copy_b128(Gh + ((size_t)t * B_ + b0) * 256 + tid * 4, gh_s + tid * 4);
            const float* Ab = Aout + ((size_t)t * B_ + b0) * 768;
            v8f zacc, tacc;
            #pragma unroll
            for (int i = 0; i < 8; ++i) {
                zacc[i] = Ab[(size_t)(i + 8 * hl) * 768 + n0 + ln];
                tacc[i] = Ab[(size_t)(i + 8 * hl) * 768 + 512 + n0 + ln];
            }
            async_wait0();
            __syncthreads();

            hd_phase(gh_s, h_lds, hdF, tid);
            __syncthreads();

            #pragma unroll
            for (int k = 0; k < 8; ++k) {
                v16bf a = *(const v16bf*)(hdF + (k * 32 + lane) * 16);
                zacc = __builtin_amdgcn_wmma_f32_16x16x32_bf16(
                    false, a, false, wz[k], (short)0, zacc, false, false);
            }
            __syncthreads();

            #pragma unroll
            for (int k = 0; k < 8; ++k) {
                v16bf a = *(const v16bf*)(rhdF + (k * 32 + lane) * 16);
                tacc = __builtin_amdgcn_wmma_f32_16x16x32_bf16(
                    false, a, false, wt[k], (short)0, tacc, false, false);
            }
            // h update (fp32 hd recomputed; each (m,n) touched by one lane)
            #pragma unroll
            for (int i = 0; i < 8; ++i) {
                int m = i + 8 * hl, n = n0 + ln;
                float hd = gh_s[m * 256 + n] * h_lds[m * 256 + n];
                float z  = sigmoidf_(zacc[i]);
                h_lds[m * 256 + n] = (1.0f - z) * hd + z * tanhf(tacc[i]);
            }
            __syncthreads();
        }
    } else {
        const int n0 = (wave - 16) * 16;
        v16bf wr[8];
        #pragma unroll
        for (int k = 0; k < 8; ++k)
            wr[k] = *(const v16bf*)(WhF + ((size_t)((1 * 8 + k) * 16 + (wave - 16)) * 32 + lane) * 16);
        __syncthreads();
        for (int t = 0; t < T_; ++t) {
            async_copy_b128(Gh + ((size_t)t * B_ + b0) * 256 + tid * 4, gh_s + tid * 4);
            const float* Ab = Aout + ((size_t)t * B_ + b0) * 768;
            v8f racc;
            #pragma unroll
            for (int i = 0; i < 8; ++i)
                racc[i] = Ab[(size_t)(i + 8 * hl) * 768 + 256 + n0 + ln];
            async_wait0();
            __syncthreads();

            hd_phase(gh_s, h_lds, hdF, tid);
            __syncthreads();

            #pragma unroll
            for (int k = 0; k < 8; ++k) {
                v16bf a = *(const v16bf*)(hdF + (k * 32 + lane) * 16);
                racc = __builtin_amdgcn_wmma_f32_16x16x32_bf16(
                    false, a, false, wr[k], (short)0, racc, false, false);
            }
            #pragma unroll
            for (int i = 0; i < 8; ++i) {
                int m = i + 8 * hl, n = n0 + ln;
                float hd = gh_s[m * 256 + n] * h_lds[m * 256 + n];
                rhdF[afrag_slot(n >> 5, m, n & 31)] = (__bf16)(sigmoidf_(racc[i]) * hd);
            }
            __syncthreads();

            // idle during the h~ GEMM: prefetch next step's gate inputs
            if (t + 1 < T_) {
                int idx = tid - 512;
                const float* AbN = Aout + ((size_t)(t + 1) * B_ + b0) * 768;
                if (idx < 384) __builtin_prefetch(AbN + idx * 32, 0, 0);
            }
            __syncthreads();
        }
    }
    __syncthreads();

    // Classifier: wave w (< 16) reduces row w with a wave32 shuffle tree
    if (wave < 16) {
        float s = 0.0f;
        #pragma unroll
        for (int i = 0; i < 8; ++i) {
            int n = lane + 32 * i;
            s += h_lds[wave * 256 + n] * Wcls[n];
        }
        #pragma unroll
        for (int off = 16; off > 0; off >>= 1) s += __shfl_xor(s, off, 32);
        if (lane == 0) out[b0 + wave] = s + bcls[0];
    }
}

// ---------------------------------------------------------------------------
extern "C" void kernel_launch(void* const* d_in, const int* in_sizes, int n_in,
                              void* d_out, int out_size, void* d_ws, size_t ws_size,
                              hipStream_t stream) {
    const float* X     = (const float*)d_in[0];
    const float* M     = (const float*)d_in[1];
    const float* Delta = (const float*)d_in[2];
    const float* xmean = (const float*)d_in[3];
    const float* Wgx   = (const float*)d_in[4];
    const float* bgx   = (const float*)d_in[5];
    const float* Wgh   = (const float*)d_in[6];
    const float* bgh   = (const float*)d_in[7];
    const float* Wx    = (const float*)d_in[8];
    const float* Wh    = (const float*)d_in[9];
    const float* Wm    = (const float*)d_in[10];
    const float* bg    = (const float*)d_in[11];
    const float* Wcls  = (const float*)d_in[12];
    const float* bcls  = (const float*)d_in[13];

    char* ws = (char*)d_ws;
    size_t off = 0;
    auto take = [&](size_t bytes) -> void* {
        void* p = ws + off;
        off = (off + bytes + 255) & ~(size_t)255;
        return p;
    };
    float*  xprev = (float*)take(sizeof(float) * (size_t)B_ * T_ * D_);          // 33.6 MB
    __bf16* XinF  = (__bf16*)take(sizeof(__bf16) * (size_t)B_ * T_ * KIN);       // 50.3 MB
    __bf16* WcatF = (__bf16*)take(sizeof(__bf16) * (size_t)6 * 64 * 512);        // 0.39 MB
    __bf16* WhF   = (__bf16*)take(sizeof(__bf16) * (size_t)3 * 8 * 16 * 512);    // 0.39 MB
    __bf16* WgxF  = (__bf16*)take(sizeof(__bf16) * (size_t)4096);                // 8 KB
    float*  Aout  = (float*)take(sizeof(float) * (size_t)B_ * T_ * 768);         // 402.7 MB
    float*  Gh    = (float*)take(sizeof(float) * (size_t)B_ * T_ * 256);         // 134.2 MB

    // 1) fragment-pack all bf16 weights
    const int npack = 6 * 64 * 512 + 3 * 8 * 16 * 512 + 4096;
    k_pack<<<(npack + 255) / 256, 256, 0, stream>>>(Wx, Wm, Wgh, Wh, Wgx,
                                                    WcatF, WhF, WgxF);
    // 2) x_prev carry scan
    k_scan<<<(B_ * D_) / 256, 256, 0, stream>>>(X, M, xprev);
    // 3) gamma_x GEMM + x_hat fragment pack (8192 waves)
    k_gammax<<<1024, 256, 0, stream>>>(X, M, Delta, xmean, WgxF, bgx, xprev, XinF);
    // 4) fused gate/gamma_h GEMM (131072 waves)
    k_biggemm<<<16384, 256, 0, stream>>>(XinF, WcatF, bg, bgh, Aout, Gh);
    // 5) serial recurrence + classifier (32 WGs, batch-partitioned)
    k_recur<<<B_ / 16, 1024, 0, stream>>>(Aout, Gh, WhF, Wcls, bcls, (float*)d_out);
}
